// GRU_2559800508940
// MI455X (gfx1250) — matrix operand order repaired and checked by reading, not compiled, not run on hardware
//
#include <hip/hip_runtime.h>

typedef _Float16 h16;
typedef __attribute__((ext_vector_type(8)))  _Float16 v8h;
typedef __attribute__((ext_vector_type(16))) _Float16 v16h;
typedef __attribute__((ext_vector_type(8)))  float    v8f;

#define NB 2048   // batch
#define NT 512    // time steps
#define NI 46     // input features
#define NH 64     // hidden
#define NC 8      // classes

__device__ __forceinline__ float rcp_fast(float x) {
#if __has_builtin(__builtin_amdgcn_rcpf)
    return __builtin_amdgcn_rcpf(x);      // v_rcp_f32
#else
    return 1.0f / x;
#endif
}
__device__ __forceinline__ float tanh_fast(float x) {
#if __has_builtin(__builtin_amdgcn_tanhf)
    return __builtin_amdgcn_tanhf(x);     // v_tanh_f32 (gfx1250 TRANS op)
#else
    float e = __expf(2.0f * x);
    return 1.0f - 2.0f * rcp_fast(e + 1.0f);
#endif
}
__device__ __forceinline__ float sigmoid_fast(float x) {
    return fmaf(0.5f, tanh_fast(0.5f * x), 0.5f);
}
__device__ __forceinline__ v16h cat8(v8h a, v8h b) {
    return __builtin_shufflevector(a, b, 0,1,2,3,4,5,6,7,8,9,10,11,12,13,14,15);
}

// Persistent fused GRU scan.
// Grid: 128 blocks (one per 16 batch rows). Block: 128 threads = 4 wave32.
// Wave j owns hidden columns [16j, 16j+16): 12x v_wmma_f32_16x16x32_f16 per
// step over A = [x(t) zero-padded to K64 | h(t-1)]. Weight B-fragments live
// in VGPRs for the whole scan; the 6 x-side WMMAs are issued BEFORE the
// per-step barrier so only the 6 h-side WMMAs + gate math are serialized.
__launch_bounds__(128, 1)
__global__ void gru_scan_kernel(const float* __restrict__ x,
                                const float* __restrict__ w_ih,
                                const float* __restrict__ w_hh,
                                const float* __restrict__ b_ih,
                                const float* __restrict__ b_hh,
                                float* __restrict__ hfin)
{
    // Double-buffered h(t) tile, 16 rows x 64 cols, padded to 72 halves/row
    // (144 B lane stride -> conflict-free ds_load_b128 across lanes 0..15).
    __shared__ h16 hbuf[2][16][72];

    const int tid   = threadIdx.x;
    const int lane  = tid & 31;
    const int wv    = tid >> 5;       // wave id == hidden-column tile j (0..3)
    const int col16 = lane & 15;
    const int hi    = lane >> 4;      // lane half (A/B fragment K split)
    const int b0    = blockIdx.x * 16;

    // ---- one-time: build this wave's 12 weight B-fragments in registers ----
    // B layout (16-bit 32x16): lanes 0-15 -> N=lane,   K = c*32 + {0..7, 16..23}
    //                          lanes16-31 -> N=lane-16, K = c*32 + {8..15, 24..31}
    // Combined K: 0..63 = w_ih (46 real, zero pad), 64..127 = w_hh.
    v16h bf[3][4];
    #pragma unroll
    for (int g = 0; g < 3; ++g) {
        const int n = g * 64 + wv * 16 + col16;   // gate row 0..191
        #pragma unroll
        for (int c = 0; c < 4; ++c) {
            v16h f;
            #pragma unroll
            for (int i = 0; i < 16; ++i) {
                int kk = c * 32 + (i < 8 ? hi * 8 + i : 16 + hi * 8 + (i - 8));
                float v;
                if (c < 2) v = (kk < NI) ? w_ih[n * NI + kk] : 0.0f;
                else       v = w_hh[n * NH + (kk - 64)];
                f[i] = (h16)v;
            }
            bf[g][c] = f;
        }
    }

    // Per-lane gate biases, pre-splatted as WMMA C operands.
    const int cr = 0   + wv * 16 + col16;
    const int cz = 64  + wv * 16 + col16;
    const int cn = 128 + wv * 16 + col16;
    const float bias_r  = b_ih[cr] + b_hh[cr];
    const float bias_z  = b_ih[cz] + b_hh[cz];
    const float bias_xn = b_ih[cn];
    const float bias_hn = b_hh[cn];
    v8f biasRv, biasZv, biasXNv, biasHNv;
    #pragma unroll
    for (int v = 0; v < 8; ++v) {
        biasRv[v] = bias_r;  biasZv[v] = bias_z;
        biasXNv[v] = bias_xn; biasHNv[v] = bias_hn;
    }

    v8h zero8;
    #pragma unroll
    for (int i = 0; i < 8; ++i) zero8[i] = (h16)0.0f;

    for (int i = tid; i < 2 * 16 * 72; i += 128) ((h16*)hbuf)[i] = (h16)0.0f;
    __syncthreads();

    const float* xrow = x + (size_t)(b0 + col16) * NT * NI;

    float hreg[8];
    #pragma unroll
    for (int v = 0; v < 8; ++v) hreg[v] = 0.0f;

    for (int t = 0; t < NT; ++t) {
        const int wb = t & 1, rb = wb ^ 1;

        // ================= pre-barrier: x(t)-only work =================
        // Prefetch a future x row into caches (global_prefetch_b8).
        int tp = (t + 8 < NT) ? (t + 8) : (NT - 1);
        __builtin_prefetch(xrow + (size_t)tp * NI, 0, 3);

        // A chunks 0,1 = x(t): rows 8B-aligned -> float2 (b64) loads.
        // A layout (16-bit 16x32): lane<16:  K {g..g+7, g+16..g+23}
        //                          lane>=16: shifted by 8.
        const float2* xr2 = (const float2*)(xrow + (size_t)t * NI);
        v16h a0, a1;
        #pragma unroll
        for (int i = 0; i < 4; ++i) {            // chunk 0: K 0..31, all real
            float2 p0 = xr2[hi * 4 + i];         // K g0+2i, g0+2i+1
            float2 p1 = xr2[8 + hi * 4 + i];     // K g1+2i, g1+2i+1
            a0[2 * i]     = (h16)p0.x;  a0[2 * i + 1]     = (h16)p0.y;
            a0[8 + 2 * i] = (h16)p1.x;  a0[8 + 2 * i + 1] = (h16)p1.y;
        }
        #pragma unroll
        for (int i = 0; i < 4; ++i) {            // chunk 1: K 32..45 real
            int k0 = 32 + hi * 8 + 2 * i;        // pair fully valid iff k0<46
            float2 p = (k0 < NI) ? xr2[k0 >> 1] : float2{0.0f, 0.0f};
            a1[2 * i] = (h16)p.x;  a1[2 * i + 1] = (h16)p.y;
        }
        {
            v8h a1lo = __builtin_shufflevector(a1, a1, 0,1,2,3,4,5,6,7);
            a1 = cat8(a1lo, zero8);              // K 48..63 always pad
        }

        // 6 x-side WMMAs (independent of h(t-1)).
        v8f accR = __builtin_amdgcn_wmma_f32_16x16x32_f16(
            false, a0, false, bf[0][0], (short)0, biasRv, false, false);
        v8f accZ = __builtin_amdgcn_wmma_f32_16x16x32_f16(
            false, a0, false, bf[1][0], (short)0, biasZv, false, false);
        v8f accXN = __builtin_amdgcn_wmma_f32_16x16x32_f16(
            false, a0, false, bf[2][0], (short)0, biasXNv, false, false);
        accR = __builtin_amdgcn_wmma_f32_16x16x32_f16(
            false, a1, false, bf[0][1], (short)0, accR, false, false);
        accZ = __builtin_amdgcn_wmma_f32_16x16x32_f16(
            false, a1, false, bf[1][1], (short)0, accZ, false, false);
        accXN = __builtin_amdgcn_wmma_f32_16x16x32_f16(
            false, a1, false, bf[2][1], (short)0, accXN, false, false);

        // ================= barrier: h(t-1) now visible =================
        __syncthreads();

        // A chunks 2,3 = h(t-1) from LDS (conflict-free b128 loads).
        v16h a2, a3;
        {
            v8h lo = *(const v8h*)&hbuf[rb][col16][hi * 8];
            v8h hh = *(const v8h*)&hbuf[rb][col16][16 + hi * 8];
            a2 = cat8(lo, hh);
            v8h lo2 = *(const v8h*)&hbuf[rb][col16][32 + hi * 8];
            v8h hh2 = *(const v8h*)&hbuf[rb][col16][48 + hi * 8];
            a3 = cat8(lo2, hh2);
        }

        // 6 h-side WMMAs (the serialized recurrence).
        v8f accHN = __builtin_amdgcn_wmma_f32_16x16x32_f16(
            false, a2, false, bf[2][2], (short)0, biasHNv, false, false);
        accR = __builtin_amdgcn_wmma_f32_16x16x32_f16(
            false, a2, false, bf[0][2], (short)0, accR, false, false);
        accZ = __builtin_amdgcn_wmma_f32_16x16x32_f16(
            false, a2, false, bf[1][2], (short)0, accZ, false, false);
        accHN = __builtin_amdgcn_wmma_f32_16x16x32_f16(
            false, a3, false, bf[2][3], (short)0, accHN, false, false);
        accR = __builtin_amdgcn_wmma_f32_16x16x32_f16(
            false, a3, false, bf[0][3], (short)0, accR, false, false);
        accZ = __builtin_amdgcn_wmma_f32_16x16x32_f16(
            false, a3, false, bf[1][3], (short)0, accZ, false, false);

        // ---- elementwise GRU update; D layout: elem v -> row v+8*hi, col lane&15
        #pragma unroll
        for (int v = 0; v < 8; ++v) {
            float r = sigmoid_fast(accR[v]);
            float z = sigmoid_fast(accZ[v]);
            float n = tanh_fast(fmaf(r, accHN[v], accXN[v]));
            float h = fmaf(z, hreg[v] - n, n);     // (1-z)*n + z*h
            hreg[v] = h;
            hbuf[wb][v + 8 * hi][wv * 16 + col16] = (h16)h;
        }
    }

    // ---- final hidden state (fp32 master copy) to workspace ----
    #pragma unroll
    for (int v = 0; v < 8; ++v) {
        int row = v + 8 * hi;
        hfin[(size_t)(b0 + row) * NH + wv * 16 + col16] = hreg[v];
    }
}

// Tiny tail: logits = h_last @ fc_w^T + fc_b, softmax over 8 classes.
__global__ void fc_softmax_kernel(const float* __restrict__ hfin,
                                  const float* __restrict__ fc_w,
                                  const float* __restrict__ fc_b,
                                  float* __restrict__ out)
{
    int b = blockIdx.x * blockDim.x + threadIdx.x;
    if (b >= NB) return;
    const float* h = hfin + (size_t)b * NH;
    float lg[NC];
    float mx = -3.0e38f;
    #pragma unroll
    for (int c = 0; c < NC; ++c) {
        float s = fc_b[c];
        const float* w = fc_w + c * NH;
        #pragma unroll 8
        for (int i = 0; i < NH; ++i) s = fmaf(h[i], w[i], s);
        lg[c] = s;
        mx = fmaxf(mx, s);
    }
    float sum = 0.0f;
    #pragma unroll
    for (int c = 0; c < NC; ++c) { lg[c] = __expf(lg[c] - mx); sum += lg[c]; }
    float inv = rcp_fast(sum);
    #pragma unroll
    for (int c = 0; c < NC; ++c) out[(size_t)b * NC + c] = lg[c] * inv;
}

extern "C" void kernel_launch(void* const* d_in, const int* in_sizes, int n_in,
                              void* d_out, int out_size, void* d_ws, size_t ws_size,
                              hipStream_t stream) {
    const float* x    = (const float*)d_in[0];
    const float* w_ih = (const float*)d_in[1];
    const float* w_hh = (const float*)d_in[2];
    const float* b_ih = (const float*)d_in[3];
    const float* b_hh = (const float*)d_in[4];
    const float* fc_w = (const float*)d_in[5];
    const float* fc_b = (const float*)d_in[6];
    float* hfin = (float*)d_ws;                 // 2048*64*4 = 512 KB scratch
    float* out  = (float*)d_out;

    gru_scan_kernel<<<NB / 16, 128, 0, stream>>>(x, w_ih, w_hh, b_ih, b_hh, hfin);
    fc_softmax_kernel<<<(NB + 255) / 256, 256, 0, stream>>>(hfin, fc_w, fc_b, out);
}
